// MultiHeadedAttentionBlock_4578435137641
// MI455X (gfx1250) — matrix operand order repaired and checked by reading, not compiled, not run on hardware
//
#include <hip/hip_runtime.h>
#include <hip/hip_bf16.h>

// MultiHeadedAttention block for MI455X (gfx1250, wave32, WMMA).
// B=2, S=2048, D=1024, H=16, HD=64. All GEMMs via v_wmma_f32_16x16x32_bf16,
// LDS-tiled + double-buffered; fragments batched into distinct registers so
// DS waits are grouped instead of per-WMMA.

#define B_  2
#define S_  2048
#define D_  1024
#define H_  16
#define HD_ 64

// padded LDS strides (elements); multiples of 8 bf16 (16B) for b128 ops,
// chosen so 16 consecutive fragment rows hit distinct bank groups.
#define LDA 40   // A/B GEMM tiles, V tile, P tile: 80B = 20 banks
#define LDK 72   // K tile: 144B = 36 banks
#define LDM 48   // mask tile (bytes)

typedef __attribute__((ext_vector_type(16))) __bf16 bf16x16;
typedef __attribute__((ext_vector_type(8)))  __bf16 bf16x8;
typedef __attribute__((ext_vector_type(8)))  float  f32x8;

static __device__ __forceinline__ bf16x16 cat8(bf16x8 lo, bf16x8 hi) {
  return __builtin_shufflevector(lo, hi, 0,1,2,3,4,5,6,7,8,9,10,11,12,13,14,15);
}

static __device__ __forceinline__ bf16x8 cvt8(float4 a, float4 b) {
  bf16x8 r;
  r[0]=(__bf16)a.x; r[1]=(__bf16)a.y; r[2]=(__bf16)a.z; r[3]=(__bf16)a.w;
  r[4]=(__bf16)b.x; r[5]=(__bf16)b.y; r[6]=(__bf16)b.z; r[7]=(__bf16)b.w;
  return r;
}

static __device__ __forceinline__ f32x8 wmma_bf16(bf16x16 a, bf16x16 b, f32x8 c) {
  return __builtin_amdgcn_wmma_f32_16x16x32_bf16(false, a, false, b, (short)0, c,
                                                 false, false);
}

static __device__ __forceinline__ bf16x16 ldsA_frag(const __bf16* base, int row) {
  const __bf16* p = base + row * LDA;           // elems 0..7 @ +0, 8..15 @ +8
  return cat8(*(const bf16x8*)(p), *(const bf16x8*)(p + 8));
}
static __device__ __forceinline__ bf16x16 ldsB_frag(const __bf16* base, int row) {
  const __bf16* p = base + row * LDA;           // 16 contiguous k
  return cat8(*(const bf16x8*)(p), *(const bf16x8*)(p + 8));
}

// ---------------------------------------------------------------------------
// Kernel 1: fused QKV projection.  qkv = q @ W_all^T  (M=B*S, N=3D, K=D)
// Block tile 128(M) x 128(N), BLK_K=32, double-buffered LDS, 8 waves as
// 4(M) x 2(N); each wave: 32x64 = 2 A-frags x 4 B-frags x 8 C-frags.
// ---------------------------------------------------------------------------
__global__ __launch_bounds__(256)
void qkv_gemm_kernel(const float* __restrict__ q,       // [B,S,D]
                     const float* __restrict__ W_all,   // [3D, D] row-major
                     __bf16* __restrict__ Qb,
                     __bf16* __restrict__ Kb,
                     __bf16* __restrict__ Vt)
{
  __shared__ __align__(16) __bf16 As[2][128 * LDA];
  __shared__ __align__(16) __bf16 Bs[2][128 * LDA];

  const int tid    = threadIdx.x;
  const int lane   = tid & 31;
  const int wave   = tid >> 5;
  const int l15    = lane & 15;
  const int hiHalf = lane >> 4;
  const int m0b = blockIdx.y * 128;
  const int n0b = blockIdx.x * 128;
  const int wm  = (wave & 3) * 32;     // wave tile origin in block tile
  const int wn  = (wave >> 2) * 64;

  const int sr = tid >> 1;             // staging row 0..127 (A and B)
  const int ss = (tid & 1) * 16;       // staging seg 0 / 16

  auto load_tile = [&](int ks, bf16x8& a0, bf16x8& a1, bf16x8& b0, bf16x8& b1) {
    const float* ap = q + (size_t)(m0b + sr) * D_ + ks + ss;
    float4 x0 = *(const float4*)(ap);
    float4 x1 = *(const float4*)(ap + 4);
    float4 x2 = *(const float4*)(ap + 8);
    float4 x3 = *(const float4*)(ap + 12);
    const float* bp = W_all + (size_t)(n0b + sr) * D_ + ks + ss;
    float4 y0 = *(const float4*)(bp);
    float4 y1 = *(const float4*)(bp + 4);
    float4 y2 = *(const float4*)(bp + 8);
    float4 y3 = *(const float4*)(bp + 12);
    a0 = cvt8(x0, x1); a1 = cvt8(x2, x3);
    b0 = cvt8(y0, y1); b1 = cvt8(y2, y3);
  };
  auto store_tile = [&](int buf, bf16x8 a0, bf16x8 a1, bf16x8 b0, bf16x8 b1) {
    *(bf16x8*)(&As[buf][sr * LDA + ss])     = a0;
    *(bf16x8*)(&As[buf][sr * LDA + ss + 8]) = a1;
    *(bf16x8*)(&Bs[buf][sr * LDA + ss])     = b0;
    *(bf16x8*)(&Bs[buf][sr * LDA + ss + 8]) = b1;
  };

  f32x8 acc[2][4] = {};
  bf16x8 sa0, sa1, sb0, sb1;
  load_tile(0, sa0, sa1, sb0, sb1);
  store_tile(0, sa0, sa1, sb0, sb1);
  __syncthreads();

  int cur = 0;
  for (int ks = 0; ks < D_; ks += 32) {
    const bool has_next = (ks + 32) < D_;
    if (has_next) load_tile(ks + 32, sa0, sa1, sb0, sb1);   // in flight

    // batch all fragment DS loads into distinct registers, then WMMA chain
    const __bf16* abase = &As[cur][(hiHalf * 16)];
    const __bf16* bbase = &Bs[cur][(hiHalf * 16)];
    bf16x16 af[2], bfv[4];
#pragma unroll
    for (int i = 0; i < 2; ++i) af[i]  = ldsA_frag(abase, wm + i*16 + l15);
#pragma unroll
    for (int t = 0; t < 4; ++t) bfv[t] = ldsB_frag(bbase, wn + t*16 + l15);
#pragma unroll
    for (int i = 0; i < 2; ++i)
#pragma unroll
      for (int t = 0; t < 4; ++t)
        acc[i][t] = wmma_bf16(af[i], bfv[t], acc[i][t]);

    if (has_next) store_tile(cur ^ 1, sa0, sa1, sb0, sb1);
    __syncthreads();
    cur ^= 1;
  }

#pragma unroll
  for (int i = 0; i < 2; ++i) {
    const int m0 = m0b + wm + i * 16;
    const int b  = m0 / S_;
    const int s0 = m0 % S_;
#pragma unroll
    for (int t = 0; t < 4; ++t) {
#pragma unroll
      for (int v = 0; v < 8; ++v) {
        const int s = s0 + v + hiHalf * 8;          // C layout: row = v (+8)
        const int n = n0b + wn + t * 16 + l15;      //           col = lane&15
        const float val = acc[i][t][v];
        if (n < D_) {                               // Q (pre-scaled 1/sqrt(64))
          const int h = n >> 6, hd = n & 63;
          Qb[(((size_t)(b*H_ + h))*S_ + s)*HD_ + hd] = (__bf16)(val * 0.125f);
        } else if (n < 2*D_) {                      // K
          const int f = n - D_, h = f >> 6, hd = f & 63;
          Kb[(((size_t)(b*H_ + h))*S_ + s)*HD_ + hd] = (__bf16)val;
        } else {                                    // V transposed
          const int f = n - 2*D_, h = f >> 6, hd = f & 63;
          Vt[(((size_t)(b*H_ + h))*HD_ + hd)*S_ + s] = (__bf16)val;
        }
      }
    }
  }
}

// ---------------------------------------------------------------------------
// Kernel 2: flash attention. One block per (b, h, 128-row q panel); all 8
// waves share the head, so K/V^T/merged-mask tiles per 32-wide kv step are
// staged once in double-buffered LDS. Fragment loads batched before WMMAs.
// ---------------------------------------------------------------------------
__global__ __launch_bounds__(256)
void attn_kernel(const __bf16* __restrict__ Qb,          // [B,H,S,HD]
                 const __bf16* __restrict__ Kb,          // [B,H,S,HD]
                 const __bf16* __restrict__ Vt,          // [B,H,HD,S]
                 const unsigned char* __restrict__ kv_mask,   // [B,S] bool
                 const unsigned char* __restrict__ attn_mask, // [B,S,S] bool
                 __bf16* __restrict__ Aout)              // [B,S,D]
{
  __shared__ __align__(16) __bf16 Ks[2][32 * LDK];            // K tile
  __shared__ __align__(16) __bf16 Vs[2][64 * LDA];            // V^T tile
  __shared__ __align__(16) unsigned char Ms[2][128 * LDM];    // merged mask
  __shared__ __align__(16) __bf16 Ps[8][16 * LDA];            // per-wave P

  const int tid    = threadIdx.x;
  const int lane   = tid & 31;
  const int wave   = tid >> 5;
  const int l15    = lane & 15;
  const int hiHalf = lane >> 4;

  const int nqblk = S_ / 128;
  const int bh  = blockIdx.x / nqblk;
  const int q0b = (blockIdx.x % nqblk) * 128;
  const int h   = bh % H_;
  const int b   = bh / H_;
  const int q0  = q0b + wave * 16;

  const __bf16* Qp = Qb + (size_t)bh * S_ * HD_;
  const __bf16* Kp = Kb + (size_t)bh * S_ * HD_;
  const __bf16* Vp = Vt + (size_t)bh * HD_ * S_;
  const unsigned char* am = attn_mask + (size_t)b * S_ * S_;
  const unsigned char* km = kv_mask   + (size_t)b * S_;

  // staging coordinates (256 threads)
  const int krow = tid >> 3, kseg = (tid & 7) * 8;   // K:   32 rows x 8 segs
  const int vrow = tid >> 2, vseg = (tid & 3) * 8;   // V^T: 64 rows x 4 segs
  const int mrow = tid >> 1, mseg = (tid & 1) * 16;  // M:  128 rows x 2 segs

  auto load_stage = [&](int kv, bf16x8& kr, bf16x8& vr, uint4& mr) {
    kr = *(const bf16x8*)(Kp + (size_t)(kv + krow) * HD_ + kseg);
    vr = *(const bf16x8*)(Vp + (size_t)vrow * S_ + kv + vseg);
    uint4 a = *(const uint4*)(am + (size_t)(q0b + mrow) * S_ + kv + mseg);
    uint4 k = *(const uint4*)(km + kv + mseg);
    mr.x = a.x & k.x; mr.y = a.y & k.y; mr.z = a.z & k.z; mr.w = a.w & k.w;
  };
  auto store_stage = [&](int buf, bf16x8 kr, bf16x8 vr, uint4 mr) {
    *(bf16x8*)(&Ks[buf][krow * LDK + kseg]) = kr;
    *(bf16x8*)(&Vs[buf][vrow * LDA + vseg]) = vr;
    *(uint4*)(&Ms[buf][mrow * LDM + mseg])  = mr;
  };

  // Q strip: two 16x32 A fragments held in VGPRs for the whole kv loop.
  bf16x16 aq[2];
  {
    const __bf16* qrow = Qp + (size_t)(q0 + l15) * HD_;
    const int koff = hiHalf * 16;
#pragma unroll
    for (int c = 0; c < 2; ++c) {
      bf16x8 lo = *(const bf16x8*)(qrow + c*32 + koff);
      bf16x8 hi = *(const bf16x8*)(qrow + c*32 + koff + 8);
      aq[c] = cat8(lo, hi);
    }
  }

  float mrowv[8], lrow[8];
  f32x8 acc[4] = {};
#pragma unroll
  for (int v = 0; v < 8; ++v) { mrowv[v] = -__builtin_inff(); lrow[v] = 0.f; }

  bf16x8 kr, vr; uint4 mr;
  load_stage(0, kr, vr, mr);
  store_stage(0, kr, vr, mr);
  __syncthreads();

  int cur = 0;
  for (int kv = 0; kv < S_; kv += 32) {
    const bool has_next = (kv + 32) < S_;
    if (has_next) {
      load_stage(kv + 32, kr, vr, mr);                 // global loads in flight
      if (kv + 64 < S_) {                              // warm L2->WGP path
        __builtin_prefetch(Kp + (size_t)(kv + 64 + krow) * HD_ + kseg);
        __builtin_prefetch(Vp + (size_t)vrow * S_ + kv + 64 + vseg);
      }
    }

    // ---- scores: batch all 4 K fragments, then 4 WMMAs
    bf16x16 kb[2][2];
#pragma unroll
    for (int t = 0; t < 2; ++t)
#pragma unroll
      for (int cc = 0; cc < 2; ++cc) {
        const __bf16* kp = &Ks[cur][(t*16 + l15) * LDK + cc*32 + hiHalf*16];
        kb[t][cc] = cat8(*(const bf16x8*)kp, *(const bf16x8*)(kp + 8));
      }
    f32x8 sfrag[2];
#pragma unroll
    for (int t = 0; t < 2; ++t) {
      f32x8 c = {};
      c = wmma_bf16(aq[0], kb[t][0], c);
      c = wmma_bf16(aq[1], kb[t][1], c);
      sfrag[t] = c;
    }

    // ---- merged mask + online softmax
    const unsigned char* mbase = &Ms[cur][0];
    float newm[8];
#pragma unroll
    for (int v = 0; v < 8; ++v) {
      const int rloc = wave * 16 + v + hiHalf * 8;
      const unsigned char* mrp = mbase + (size_t)rloc * LDM;
      const float s0 = mrp[l15]      ? sfrag[0][v] : -__builtin_inff();
      const float s1 = mrp[l15 + 16] ? sfrag[1][v] : -__builtin_inff();
      sfrag[0][v] = s0; sfrag[1][v] = s1;
      float mx = fmaxf(s0, s1);
#pragma unroll
      for (int d = 1; d < 16; d <<= 1)                 // within 16-lane half
        mx = fmaxf(mx, __shfl_xor(mx, d, 32));
      newm[v] = fmaxf(mrowv[v], mx);
    }
#pragma unroll
    for (int v = 0; v < 8; ++v) {
      const float nm = newm[v];
      const float p0 = (sfrag[0][v] == -__builtin_inff()) ? 0.f
                                                          : __expf(sfrag[0][v] - nm);
      const float p1 = (sfrag[1][v] == -__builtin_inff()) ? 0.f
                                                          : __expf(sfrag[1][v] - nm);
      sfrag[0][v] = p0; sfrag[1][v] = p1;
      float rs = p0 + p1;
#pragma unroll
      for (int d = 1; d < 16; d <<= 1)
        rs += __shfl_xor(rs, d, 32);
      const float alpha = (nm == -__builtin_inff()) ? 0.f : __expf(mrowv[v] - nm);
      lrow[v] = lrow[v] * alpha + rs;
      mrowv[v] = nm;
#pragma unroll
      for (int c = 0; c < 4; ++c) acc[c][v] *= alpha;
    }

    // ---- P: C-fragment -> per-wave LDS tile -> A-fragment (same-wave DS
    // ops are in order; tile is private, no barrier needed).
    __bf16* pt = Ps[wave];
#pragma unroll
    for (int v = 0; v < 8; ++v) {
      const int row = v + hiHalf * 8;
      pt[row * LDA + l15]      = (__bf16)sfrag[0][v];
      pt[row * LDA + 16 + l15] = (__bf16)sfrag[1][v];
    }
    bf16x16 ap;
    {
      const __bf16* prow = pt + (size_t)l15 * LDA + hiHalf * 16;
      ap = cat8(*(const bf16x8*)(prow), *(const bf16x8*)(prow + 8));
    }

    // ---- acc += P @ V: batch all 4 V fragments, then 4 WMMAs
    bf16x16 vb[4];
#pragma unroll
    for (int c = 0; c < 4; ++c) {
      const __bf16* vp = &Vs[cur][(c*16 + l15) * LDA + hiHalf * 16];
      vb[c] = cat8(*(const bf16x8*)vp, *(const bf16x8*)(vp + 8));
    }
#pragma unroll
    for (int c = 0; c < 4; ++c)
      acc[c] = wmma_bf16(ap, vb[c], acc[c]);

    if (has_next) store_stage(cur ^ 1, kr, vr, mr);
    __syncthreads();
    cur ^= 1;
  }

  // ---- finalize: a_out = acc / l (fully-masked row: l==0 -> 0, nan_to_num)
#pragma unroll
  for (int v = 0; v < 8; ++v) {
    const float inv = (lrow[v] > 0.f) ? (1.0f / lrow[v]) : 0.f;
    const int s = q0 + v + hiHalf * 8;
    __bf16* orow = Aout + ((size_t)b * S_ + s) * D_ + h * HD_;
#pragma unroll
    for (int c = 0; c < 4; ++c)
      orow[c*16 + l15] = (__bf16)(acc[c][v] * inv);
  }
}

// ---------------------------------------------------------------------------
// Kernel 3: output projection.  out = a_out @ W_out^T + b_out  (fp32 out)
// Same 128x128 LDS-tiled scheme; A is already bf16.
// ---------------------------------------------------------------------------
__global__ __launch_bounds__(256)
void out_gemm_kernel(const __bf16* __restrict__ Aout,   // [B*S, D] bf16
                     const float* __restrict__ W_out,   // [D, D] row-major
                     const float* __restrict__ b_out,   // [D]
                     float* __restrict__ out)           // [B,S,D]
{
  __shared__ __align__(16) __bf16 As[2][128 * LDA];
  __shared__ __align__(16) __bf16 Bs[2][128 * LDA];

  const int tid    = threadIdx.x;
  const int lane   = tid & 31;
  const int wave   = tid >> 5;
  const int l15    = lane & 15;
  const int hiHalf = lane >> 4;
  const int m0b = blockIdx.y * 128;
  const int n0b = blockIdx.x * 128;
  const int wm  = (wave & 3) * 32;
  const int wn  = (wave >> 2) * 64;

  const int sr = tid >> 1;
  const int ss = (tid & 1) * 16;

  auto load_tile = [&](int ks, bf16x8& a0, bf16x8& a1, bf16x8& b0, bf16x8& b1) {
    const __bf16* ap = Aout + (size_t)(m0b + sr) * D_ + ks + ss;
    a0 = *(const bf16x8*)(ap);
    a1 = *(const bf16x8*)(ap + 8);
    const float* bp = W_out + (size_t)(n0b + sr) * D_ + ks + ss;
    float4 y0 = *(const float4*)(bp);
    float4 y1 = *(const float4*)(bp + 4);
    float4 y2 = *(const float4*)(bp + 8);
    float4 y3 = *(const float4*)(bp + 12);
    b0 = cvt8(y0, y1); b1 = cvt8(y2, y3);
  };
  auto store_tile = [&](int buf, bf16x8 a0, bf16x8 a1, bf16x8 b0, bf16x8 b1) {
    *(bf16x8*)(&As[buf][sr * LDA + ss])     = a0;
    *(bf16x8*)(&As[buf][sr * LDA + ss + 8]) = a1;
    *(bf16x8*)(&Bs[buf][sr * LDA + ss])     = b0;
    *(bf16x8*)(&Bs[buf][sr * LDA + ss + 8]) = b1;
  };

  f32x8 acc[2][4] = {};
  bf16x8 sa0, sa1, sb0, sb1;
  load_tile(0, sa0, sa1, sb0, sb1);
  store_tile(0, sa0, sa1, sb0, sb1);
  __syncthreads();

  int cur = 0;
  for (int ks = 0; ks < D_; ks += 32) {
    const bool has_next = (ks + 32) < D_;
    if (has_next) load_tile(ks + 32, sa0, sa1, sb0, sb1);

    const __bf16* abase = &As[cur][(hiHalf * 16)];
    const __bf16* bbase = &Bs[cur][(hiHalf * 16)];
    bf16x16 af[2], bfv[4];
#pragma unroll
    for (int i = 0; i < 2; ++i) af[i]  = ldsA_frag(abase, wm + i*16 + l15);
#pragma unroll
    for (int t = 0; t < 4; ++t) bfv[t] = ldsB_frag(bbase, wn + t*16 + l15);
#pragma unroll
    for (int i = 0; i < 2; ++i)
#pragma unroll
      for (int t = 0; t < 4; ++t)
        acc[i][t] = wmma_bf16(af[i], bfv[t], acc[i][t]);

    if (has_next) store_tile(cur ^ 1, sa0, sa1, sb0, sb1);
    __syncthreads();
    cur ^= 1;
  }

#pragma unroll
  for (int i = 0; i < 2; ++i) {
#pragma unroll
    for (int t = 0; t < 4; ++t) {
      const int n = n0b + wn + t * 16 + l15;
      const float bias = b_out[n];
#pragma unroll
      for (int v = 0; v < 8; ++v) {
        const int m = m0b + wm + i * 16 + v + hiHalf * 8;
        out[(size_t)m * D_ + n] = acc[i][t][v] + bias;
      }
    }
  }
}

// ---------------------------------------------------------------------------
extern "C" void kernel_launch(void* const* d_in, const int* in_sizes, int n_in,
                              void* d_out, int out_size, void* d_ws, size_t ws_size,
                              hipStream_t stream) {
  const float*         q         = (const float*)d_in[0];
  const unsigned char* kv_mask   = (const unsigned char*)d_in[1]; // bool [B,S]
  const unsigned char* attn_mask = (const unsigned char*)d_in[2]; // bool [B,S,S]
  const float*         W_all     = (const float*)d_in[3];
  const float*         W_out     = (const float*)d_in[4];
  const float*         b_out     = (const float*)d_in[5];
  float*               out       = (float*)d_out;

  // Workspace carve-out: 4 x 8 MB bf16 buffers (32 MB total).
  char* ws = (char*)d_ws;
  const size_t nQ = (size_t)B_ * H_ * S_ * HD_;   // == B*S*D
  __bf16* Qb   = (__bf16*)ws;              ws += nQ * sizeof(__bf16);
  __bf16* Kb   = (__bf16*)ws;              ws += nQ * sizeof(__bf16);
  __bf16* Vt   = (__bf16*)ws;              ws += nQ * sizeof(__bf16);
  __bf16* Aout = (__bf16*)ws;

  const dim3 blk(256);
  qkv_gemm_kernel<<<dim3(3*D_/128, (B_*S_)/128), blk, 0, stream>>>(q, W_all, Qb, Kb, Vt);
  attn_kernel<<<dim3(B_*H_*(S_/128)), blk, 0, stream>>>(Qb, Kb, Vt, kv_mask,
                                                        attn_mask, Aout);
  out_gemm_kernel<<<dim3(D_/128, (B_*S_)/128), blk, 0, stream>>>(Aout, W_out, b_out, out);
}
